// FusionMamba_54065048322126
// MI455X (gfx1250) — matrix operand order repaired and checked by reading, not compiled, not run on hardware
//
#include <hip/hip_runtime.h>
#include <hip/hip_bf16.h>
#include <cstdint>

typedef _Float16 h16;
typedef __attribute__((ext_vector_type(16))) _Float16 v16h;
typedef __attribute__((ext_vector_type(8)))  float    v8f;

// ---------------- WMMA fragment helpers (wave32, 16x16x32 f16) ----------------
// A-matrix 16x32 f16 fragment (this code's contiguous k order): lane = half*16 + m(0..15);
// element e: k = half*16 + e (matches B fragment)
__device__ __forceinline__ v16h load_a_frag(const h16* base, int row0, int stride, int k0) {
  const int lane = threadIdx.x & 31;
  const int hh = lane >> 4;
  const h16* p = base + (size_t)(row0 + (lane & 15)) * stride + k0;
  v16h f;
#pragma unroll
  for (int e = 0; e < 16; ++e) {
    int kk = hh * 16 + e;
    f[e] = p[kk];
  }
  return f;
}

// B-matrix 32x16 f16: lane = half*16 + n; element e: k = half*16 + e.
// B[k][n] = W[n][k] for row-major weight W (N x K).
__device__ __forceinline__ v16h load_b_frag(const h16* base, int col0, int stride, int k0) {
  const int lane = threadIdx.x & 31;
  const int hh = lane >> 4;
  const h16* p = base + (size_t)(col0 + (lane & 15)) * stride + k0 + hh * 16;
  v16h f;
#pragma unroll
  for (int e = 0; e < 16; ++e) f[e] = p[e];
  return f;
}

__device__ __forceinline__ v8f wmma_f16(v16h a, v16h b, v8f c) {
  return __builtin_amdgcn_wmma_f32_16x16x32_f16(false, a, false, b, (short)0, c, false, false);
}

// CDNA5 async memory->LDS copy (VGLOBAL GLOBAL_LOAD_ASYNC_TO_LDS_B64, ASYNCcnt)
__device__ __forceinline__ void async_copy_b64(uint32_t lds_off, const void* gaddr) {
  asm volatile("global_load_async_to_lds_b64 %0, %1, off"
               :: "v"(lds_off), "v"((uint64_t)(uintptr_t)gaddr)
               : "memory");
}
__device__ __forceinline__ void wait_async0() {
  asm volatile("s_wait_asynccnt 0" ::: "memory");
}

__device__ __forceinline__ float siluf(float x) {
  return x * __builtin_amdgcn_rcpf(1.0f + __expf(-x));
}
__device__ __forceinline__ float softplusf(float x) {
  return (x > 20.0f) ? x : __logf(1.0f + __expf(x));
}

// cross-scan position permutation: internal index m of direction k -> index into X(B,L,128)
__device__ __forceinline__ int permPos(int k, int m) {
  int mm = (k & 2) ? (4095 - m) : m;
  if (k & 1) mm = ((mm & 63) << 6) | (mm >> 6);
  return mm;
}

// ---------------- kernel 0: weight prep (f32 -> f16, A = -exp(A_logs)) --------
__global__ void k_prep(const float* pw, const float* xw, const float* ow,
                       const float* f1, const float* f2, const float* alog,
                       h16* pwh, h16* xwh, h16* owh, h16* f1h, h16* f2h, float* An) {
  for (int i = blockIdx.x * blockDim.x + threadIdx.x; i < 262144;
       i += gridDim.x * blockDim.x) {
    if (i < 32768)  pwh[i] = (h16)pw[i];                       // (2,256,64)
    if (i < 24576) {                                           // (4,48,128) padded
      int k = i / (48 * 128), rem = i % (48 * 128);
      int j = rem / 128, d = rem % 128;
      xwh[i] = (j < 36) ? (h16)xw[(k * 36 + j) * 128 + d] : (h16)0.0f;
    }
    if (i < 8192)   owh[i] = (h16)ow[i];                       // (64,128)
    if (i < 16384)  f1h[i] = (h16)f1[i];                       // (256,64)
    if (i < 196608) f2h[i] = (h16)f2[i];                       // (768,256)
    if (i < 8192)   An[i]  = -__expf(alog[i]);                 // (512,16)
  }
}

// ---------------- kernel 1: in-LN + dual projection + gate ------------------
// grid 4096 blocks of 256 (16 pixel-rows per block)
// X  (B*L,128) f32 = p0+p1 ; Zc (B*L,128) f16 = silu(z0)+silu(z1)
__global__ void k_inproj(const float* __restrict__ x0, const float* __restrict__ x1,
                         const float* __restrict__ g, const float* __restrict__ bta,
                         const h16* __restrict__ pwh,
                         float* __restrict__ X, h16* __restrict__ Zc) {
  __shared__ h16 Aln[2][16][64];
  const int t = threadIdx.x;
  const int r0 = blockIdx.x * 16;

  if (t < 32) {
    int mod = t & 1, pix = t >> 1;
    int r = r0 + pix, b = r >> 12, l = r & 4095;
    const float* src = (mod ? x1 : x0) + (size_t)b * 64 * 4096 + l;
    float m = 0.f;
#pragma unroll 8
    for (int c = 0; c < 64; ++c) m += src[(size_t)c * 4096];
    m *= (1.0f / 64.0f);
    float v = 0.f;
#pragma unroll 8
    for (int c = 0; c < 64; ++c) { float d = src[(size_t)c * 4096] - m; v += d * d; }
    v *= (1.0f / 64.0f);
    float rs = rsqrtf(v + 1e-5f);
#pragma unroll 8
    for (int c = 0; c < 64; ++c)
      Aln[mod][pix][c] = (h16)((src[(size_t)c * 4096] - m) * rs * g[c] + bta[c]);
  }
  __syncthreads();

  const int wave = t >> 5, lane = t & 31, hh = lane >> 4;
#pragma unroll
  for (int j = 0; j < 2; ++j) {
    int nt = wave + 8 * j;                   // 0..15 (cols nt*16..)
    v8f acc0 = {}, acc1 = {};
#pragma unroll
    for (int ks = 0; ks < 2; ++ks) {
      v16h a0 = load_a_frag(&Aln[0][0][0], 0, 64, ks * 32);
      v16h b0 = load_b_frag(pwh, nt * 16, 64, ks * 32);
      acc0 = wmma_f16(a0, b0, acc0);
      v16h a1 = load_a_frag(&Aln[1][0][0], 0, 64, ks * 32);
      v16h b1 = load_b_frag(pwh + 16384, nt * 16, 64, ks * 32);
      acc1 = wmma_f16(a1, b1, acc1);
    }
#pragma unroll
    for (int v = 0; v < 8; ++v) {
      int r = r0 + v + 8 * hh;
      int col = nt * 16 + (lane & 15);
      if (nt < 8) X[(size_t)r * 128 + col] = acc0[v] + acc1[v];
      else        Zc[(size_t)r * 128 + (col - 128)] = (h16)(siluf(acc0[v]) + siluf(acc1[v]));
    }
  }
}

// ---------------- kernel 2: fused x_dbl GEMM + selective scan ----------------
// 64 blocks (b,k), 256 threads; chunked: WMMA projection -> LDS -> recurrence
__global__ void k_scan(const float* __restrict__ X, const h16* __restrict__ xwh,
                       const float* __restrict__ dtw, const float* __restrict__ dtb,
                       const float* __restrict__ An, const float* __restrict__ Dsv,
                       float* __restrict__ Y4) {
  __shared__ h16   Wx[48 * 128];   // x_proj (padded) f16
  __shared__ float Wdt[128 * 4];
  __shared__ float biasS[128], Dss[128];
  __shared__ float Asm[128 * 16];
  __shared__ h16   Xc[32 * 128];   // chunk of inputs (f16)
  __shared__ float Cc[32 * 48];    // x_dbl outputs: [l][0:4]=dt_rank [4:20]=B [20:36]=C
  __shared__ float Dl[32 * 128];   // delta

  const int t = threadIdx.x;
  const int bk = blockIdx.x, b = bk >> 2, k = bk & 3;
  const int wave = t >> 5, lane = t & 31, hh = lane >> 4;

  for (int i = t; i < 48 * 128; i += 256) Wx[i] = xwh[k * 48 * 128 + i];
  for (int i = t; i < 512; i += 256) Wdt[i] = dtw[k * 512 + i];
  if (t < 128) { biasS[t] = dtb[k * 128 + t]; Dss[t] = Dsv[k * 128 + t]; }
  for (int i = t; i < 2048; i += 256) Asm[i] = An[k * 2048 + i];
  __syncthreads();

  const int d = t >> 1, nq = t & 1, n0 = nq * 8;
  float h[8], Arow[8];
#pragma unroll
  for (int i = 0; i < 8; ++i) { h[i] = 0.f; Arow[i] = Asm[d * 16 + n0 + i]; }
  const float Dd = Dss[d];

  for (int ch = 0; ch < 128; ++ch) {
    // stage 32 permuted rows of X as f16
#pragma unroll
    for (int j = 0; j < 16; ++j) {
      int idx = t + j * 256;
      int li = idx >> 7, dd = idx & 127;
      int p = permPos(k, ch * 32 + li);
      Xc[li * 128 + dd] = (h16)X[((size_t)b * 4096 + p) * 128 + dd];
    }
    __syncthreads();

    // c[32x48] = Xc(32x128) * Wx^T(48x128) via WMMA (2 m-tiles x 3 n-tiles)
    if (wave < 6) {
      int mt = wave / 3, nt = wave % 3;
      v8f acc = {};
#pragma unroll
      for (int ks = 0; ks < 4; ++ks) {
        v16h a = load_a_frag(Xc, mt * 16, 128, ks * 32);
        v16h bf = load_b_frag(Wx, nt * 16, 128, ks * 32);
        acc = wmma_f16(a, bf, acc);
      }
#pragma unroll
      for (int v = 0; v < 8; ++v)
        Cc[(mt * 16 + v + 8 * hh) * 48 + nt * 16 + (lane & 15)] = acc[v];
    }
    __syncthreads();

    // delta = softplus(dt_proj(c[:,:4]) + bias)
#pragma unroll
    for (int j = 0; j < 16; ++j) {
      int idx = t + j * 256;
      int li = idx >> 7, dd = idx & 127;
      float dt = biasS[dd];
#pragma unroll
      for (int r = 0; r < 4; ++r) dt += Cc[li * 48 + r] * Wdt[dd * 4 + r];
      Dl[li * 128 + dd] = softplusf(dt);
    }
    __syncthreads();

    // sequential recurrence over the chunk
    for (int li = 0; li < 32; ++li) {
      float delta = Dl[li * 128 + d];
      float x = (float)Xc[li * 128 + d];
      float dx = delta * x;
      float yp = 0.f;
#pragma unroll
      for (int i = 0; i < 8; ++i) {
        int n = n0 + i;
        float Bn = Cc[li * 48 + 4 + n];
        float Cn = Cc[li * 48 + 20 + n];
        float dA = __expf(delta * Arow[i]);
        h[i] = dA * h[i] + dx * Bn;
        yp += h[i] * Cn;
      }
      yp += __shfl_xor(yp, 1, 32);
      if (nq == 0)
        Y4[((size_t)bk * 4096 + ch * 32 + li) * 128 + d] = yp + Dd * x;
    }
    __syncthreads();
  }
}

// ---------------- kernel 3: combine 4 dirs + out-LN + gate ------------------
__global__ void k_combine(const float* __restrict__ Y4, const h16* __restrict__ Zc,
                          const float* __restrict__ g, const float* __restrict__ bt,
                          h16* __restrict__ Yg) {
  __shared__ float red[8];
  const int r = blockIdx.x, b = r >> 12, l = r & 4095;
  const int d = threadIdx.x, w = d >> 5, lane = d & 31;
  int p1 = ((l & 63) << 6) | (l >> 6);
  float v = Y4[(((size_t)b * 4 + 0) * 4096 + l) * 128 + d]
          + Y4[(((size_t)b * 4 + 2) * 4096 + (4095 - l)) * 128 + d]
          + Y4[(((size_t)b * 4 + 1) * 4096 + p1) * 128 + d]
          + Y4[(((size_t)b * 4 + 3) * 4096 + (4095 - p1)) * 128 + d];
  float s = v;
#pragma unroll
  for (int o = 16; o > 0; o >>= 1) s += __shfl_xor(s, o, 32);
  if (lane == 0) red[w] = s;
  __syncthreads();
  float mean = (red[0] + red[1] + red[2] + red[3]) * (1.0f / 128.0f);
  __syncthreads();
  float dv = v - mean, s2 = dv * dv;
#pragma unroll
  for (int o = 16; o > 0; o >>= 1) s2 += __shfl_xor(s2, o, 32);
  if (lane == 0) red[w] = s2;
  __syncthreads();
  float var = (red[0] + red[1] + red[2] + red[3]) * (1.0f / 128.0f);
  float yln = dv * rsqrtf(var + 1e-5f) * g[d] + bt[d];
  Yg[(size_t)r * 128 + d] = (h16)(yln * (float)Zc[(size_t)r * 128 + d]);
}

// ---------------- kernel 4: generic WMMA GEMM with fused epilogues ----------
// C(M,N) = A(M,K)f16 @ W(N,K)^T ; block = 4 waves, 16 rows x 64 cols per block
// A tiles staged memory->LDS with GLOBAL_LOAD_ASYNC_TO_LDS_B64 (ASYNCcnt).
// mode 1: +residual(x0+x1, NCHW) -> f16 ; mode 2: +bias, exact GELU -> f16 ; mode 3: +bias -> f32
__global__ void k_gemm(const h16* __restrict__ A, const h16* __restrict__ W,
                       const float* __restrict__ bias, int K, int N, int mode,
                       const float* __restrict__ res0, const float* __restrict__ res1,
                       h16* __restrict__ outh, float* __restrict__ outf) {
  __shared__ h16 As[16 * 32];
  const int t = threadIdx.x, wave = t >> 5, lane = t & 31, hh = lane >> 4;
  const int r0 = blockIdx.x * 16;
  const int n0 = (blockIdx.y * 4 + wave) * 16;
  const int ksteps = K >> 5;
  // each thread owns 4 consecutive halves (8B) of the staged 16x32 tile
  const uint32_t lds_off = (uint32_t)(uintptr_t)&As[t * 4];
  const h16* arow = A + (size_t)(r0 + (t >> 3)) * K + (t & 7) * 4;
  v8f acc = {};
  for (int ks = 0; ks < ksteps; ++ks) {
    async_copy_b64(lds_off, arow + ks * 32);
    if (ks + 1 < ksteps)
      __builtin_prefetch(arow + (ks + 1) * 32, 0, 1);
    wait_async0();
    __syncthreads();
    v16h a = load_a_frag(As, 0, 32, 0);
    v16h bf = load_b_frag(W, n0, K, ks * 32);
    acc = wmma_f16(a, bf, acc);
    __syncthreads();
  }
#pragma unroll
  for (int v = 0; v < 8; ++v) {
    int r = r0 + v + 8 * hh;
    int n = n0 + (lane & 15);
    float val = acc[v];
    if (mode == 1) {
      int b = r >> 12, l = r & 4095;
      size_t ri = ((size_t)b * 64 + n) * 4096 + l;
      val += res0[ri] + res1[ri];
      outh[(size_t)r * N + n] = (h16)val;
    } else if (mode == 2) {
      val += bias[n];
      val = 0.5f * val * (1.0f + erff(val * 0.70710678118654752f));
      outh[(size_t)r * N + n] = (h16)val;
    } else {
      val += bias[n];
      outf[(size_t)r * N + n] = val;
    }
  }
}

// ---------------- kernel 5: final in-place LN over 768 ----------------------
__global__ void k_final_ln(float* __restrict__ out, const float* __restrict__ g,
                           const float* __restrict__ bt) {
  __shared__ float red[8];
  const int t = threadIdx.x, w = t >> 5, lane = t & 31;
  float* row = out + (size_t)blockIdx.x * 768;
  float x[3], s = 0.f;
#pragma unroll
  for (int i = 0; i < 3; ++i) { x[i] = row[t + i * 256]; s += x[i]; }
#pragma unroll
  for (int o = 16; o > 0; o >>= 1) s += __shfl_xor(s, o, 32);
  if (lane == 0) red[w] = s;
  __syncthreads();
  float mean = 0.f;
#pragma unroll
  for (int i = 0; i < 8; ++i) mean += red[i];
  mean *= (1.0f / 768.0f);
  __syncthreads();
  float s2 = 0.f;
#pragma unroll
  for (int i = 0; i < 3; ++i) { float dv = x[i] - mean; s2 += dv * dv; }
#pragma unroll
  for (int o = 16; o > 0; o >>= 1) s2 += __shfl_xor(s2, o, 32);
  if (lane == 0) red[w] = s2;
  __syncthreads();
  float var = 0.f;
#pragma unroll
  for (int i = 0; i < 8; ++i) var += red[i];
  var *= (1.0f / 768.0f);
  float rs = rsqrtf(var + 1e-5f);
#pragma unroll
  for (int i = 0; i < 3; ++i) {
    int c = t + i * 256;
    row[c] = (x[i] - mean) * rs * g[c] + bt[c];
  }
}

// ---------------- launcher --------------------------------------------------
extern "C" void kernel_launch(void* const* d_in, const int* in_sizes, int n_in,
                              void* d_out, int out_size, void* d_ws, size_t ws_size,
                              hipStream_t stream) {
  (void)in_sizes; (void)n_in; (void)out_size; (void)ws_size;
  const float* x0  = (const float*)d_in[0];
  const float* x1  = (const float*)d_in[1];
  const float* ing = (const float*)d_in[2];
  const float* inb = (const float*)d_in[3];
  const float* pw  = (const float*)d_in[4];
  const float* xw  = (const float*)d_in[5];
  const float* dtw = (const float*)d_in[6];
  const float* alog= (const float*)d_in[7];
  const float* Dsv = (const float*)d_in[8];
  const float* ong = (const float*)d_in[9];
  const float* onb = (const float*)d_in[10];
  const float* ow  = (const float*)d_in[11];
  const float* f1w = (const float*)d_in[12];
  const float* f1b = (const float*)d_in[13];
  const float* f2w = (const float*)d_in[14];
  const float* f2b = (const float*)d_in[15];
  const float* lng = (const float*)d_in[16];
  const float* lnb = (const float*)d_in[17];
  const float* dtb = (const float*)d_in[18];
  float* out = (float*)d_out;

  char* ws = (char*)d_ws;
  size_t off = 0;
  auto alloc = [&](size_t bytes) -> void* {
    void* p = ws + off; off += (bytes + 255) & ~(size_t)255; return p;
  };
  float* X   = (float*)alloc(65536ull * 128 * 4);
  h16*   Zc  = (h16*)  alloc(65536ull * 128 * 2);
  float* Y4  = (float*)alloc(64ull * 4096 * 128 * 4);
  h16*   Yg  = (h16*)  alloc(65536ull * 128 * 2);
  h16*   Fh  = (h16*)  alloc(65536ull * 64 * 2);
  h16*   H1  = (h16*)  alloc(65536ull * 256 * 2);
  h16*   pwh = (h16*)  alloc(32768 * 2);
  h16*   xwh = (h16*)  alloc(24576 * 2);
  h16*   owh = (h16*)  alloc(8192 * 2);
  h16*   f1h = (h16*)  alloc(16384 * 2);
  h16*   f2h = (h16*)  alloc(196608 * 2);
  float* An  = (float*)alloc(8192 * 4);

  k_prep<<<512, 256, 0, stream>>>(pw, xw, ow, f1w, f2w, alog,
                                  pwh, xwh, owh, f1h, f2h, An);
  k_inproj<<<4096, 256, 0, stream>>>(x0, x1, ing, inb, pwh, X, Zc);
  k_scan<<<64, 256, 0, stream>>>(X, xwh, dtw, dtb, An, Dsv, Y4);
  k_combine<<<65536, 128, 0, stream>>>(Y4, Zc, ong, onb, Yg);
  // out_proj + residual
  k_gemm<<<dim3(4096, 1), 128, 0, stream>>>(Yg, owh, nullptr, 128, 64, 1, x0, x1, Fh, nullptr);
  // fc1 + GELU
  k_gemm<<<dim3(4096, 4), 128, 0, stream>>>(Fh, f1h, f1b, 64, 256, 2, nullptr, nullptr, H1, nullptr);
  // fc2 + bias -> d_out
  k_gemm<<<dim3(4096, 12), 128, 0, stream>>>(H1, f2h, f2b, 256, 768, 3, nullptr, nullptr, nullptr, out);
  k_final_ln<<<65536, 256, 0, stream>>>(out, lng, lnb);
}
